// Llama4VisionAttention_1614907703561
// MI455X (gfx1250) — matrix operand order repaired and checked by reading, not compiled, not run on hardware
//
#include <hip/hip_runtime.h>

// ---------------------------------------------------------------------------
// Llama4 vision attention forward for gfx1250 (MI455X), bf16 WMMA everywhere,
// register-blocked for L2-bandwidth reuse.
// Workspace layout (~156 MB): xb | Wq,Wk,Wv,Wo (bf16) | Qb | Kb | Vt | attn
// ---------------------------------------------------------------------------

typedef __attribute__((ext_vector_type(16))) __bf16 v16bf;
typedef __attribute__((ext_vector_type(8)))  float  v8f;

struct alignas(16) U4 { unsigned int x, y, z, w; };

union FragU {
  U4           h[2];
  unsigned int u[8];
  v16bf        v;
};
static_assert(sizeof(FragU) == 32, "frag size");

constexpr int Bc  = 16;
constexpr int Sc  = 577;
constexpr int Dc  = 1408;
constexpr int Hc  = 16;
constexpr int HDc = 88;
constexpr int HD2c = 44;
constexpr int Mc  = Bc * Sc;      // 9232 = 577 full M-tiles of 16
constexpr int SPc  = 592;         // S padded to 37*16 (query rows)
constexpr int SKPc = 608;         // S padded to 19*32 (key K-dim)
constexpr int HDPc = 96;          // head_dim padded to 3*32
constexpr int KT = Dc / 32;       // 44
constexpr int NT = Dc / 16;       // 88
constexpr int MT = Mc / 16;       // 577
constexpr int QTILES = SPc / 16;  // 37
constexpr int KBLK   = SKPc / 32; // 19

constexpr int MB  = 4;                      // M-tiles/wave in QKV gemm
constexpr int OMB = 4, ONB = 4;             // M/N-tiles/wave in out-proj
constexpr int QW  = 2;                      // query tiles/wave in attention
constexpr int MBLK  = (MT + MB - 1) / MB;       // 145
constexpr int OMBLK = (MT + OMB - 1) / OMB;     // 145
constexpr int ONBLK = NT / ONB;                 // 22
constexpr int QBLK  = (QTILES + QW - 1) / QW;   // 19

constexpr size_t XB_BYTES  = (size_t)Mc * Dc * 2;
constexpr size_t W_BYTES   = (size_t)Dc * Dc * 2;
constexpr size_t QKB_BYTES = (size_t)Bc * Hc * SPc * HDPc * 2;
constexpr size_t VT_BYTES  = (size_t)Bc * Hc * HDPc * SKPc * 2;
constexpr size_t AT_BYTES  = XB_BYTES;

__device__ __forceinline__ int imin(int a, int b) { return a < b ? a : b; }

__device__ __forceinline__ unsigned short f2bf(float x) {
  unsigned int u = __float_as_uint(x);
  u += 0x7fffu + ((u >> 16) & 1u);           // round-to-nearest-even
  return (unsigned short)(u >> 16);
}
__device__ __forceinline__ unsigned int pack2bf(float a, float b) {
  return (unsigned int)f2bf(a) | ((unsigned int)f2bf(b) << 16);
}
__device__ __forceinline__ v8f wmma_bf16(v16bf a, v16bf b, v8f c) {
  return __builtin_amdgcn_wmma_f32_16x16x32_bf16(false, a, false, b,
                                                 (short)0, c, false, false);
}

// ---------------------------------------------------------------------------
__global__ __launch_bounds__(256) void cvt_f32_bf16_x4(
    const float4* __restrict__ s, uint2* __restrict__ d, int n4) {
  int i = blockIdx.x * blockDim.x + threadIdx.x;
  if (i < n4) {
    float4 f = s[i];
    uint2 o;
    o.x = pack2bf(f.x, f.y);
    o.y = pack2bf(f.z, f.w);
    d[i] = o;
  }
}

// ---------------------------------------------------------------------------
// Fused QKV GEMM + bias + RoPE epilogue.
// One wave computes MB(=4) M-tiles x 1 N-tile for q, k and v simultaneously:
// the 3 weight B-fragments are reused across the 4 A-fragments (28 FLOP/B).
__global__ __launch_bounds__(32) void qkv_rope_kernel(
    const unsigned short* __restrict__ xb,
    const unsigned short* __restrict__ wqb,
    const unsigned short* __restrict__ wkb,
    const unsigned short* __restrict__ wvb,
    const float* __restrict__ bq, const float* __restrict__ bk,
    const float* __restrict__ bv,
    const float* __restrict__ fcos, const float* __restrict__ fsin,
    unsigned short* __restrict__ qbuf,   // [B*H, 592, 96]
    unsigned short* __restrict__ kbuf,   // [B*H, 592, 96]
    unsigned short* __restrict__ vtb)    // [B*H, 96, 608]
{
  const int nt  = blockIdx.x;            // 0..87
  const int mtb = blockIdx.y * MB;       // first M-tile of this wave
  const int lane = threadIdx.x;
  const int hv  = lane >> 4;             // half-wave
  const int lid = lane & 15;

  const int d = nt * 16 + lid;           // output feature (B frag / C column)
  const unsigned short* wq = wqb + (size_t)d * Dc;
  const unsigned short* wk = wkb + (size_t)d * Dc;
  const unsigned short* wv = wvb + (size_t)d * Dc;

  const unsigned short* xr[MB];
#pragma unroll
  for (int i = 0; i < MB; ++i) {
    const int row = imin((mtb + i) * 16 + lid, Mc - 1);  // clamp tail block
    xr[i] = xb + (size_t)row * Dc;
  }

  v8f cq[MB], ck[MB], cv[MB];
#pragma unroll
  for (int i = 0; i < MB; ++i) { cq[i] = {}; ck[i] = {}; cv[i] = {}; }

  for (int kt = 0; kt < KT; ++kt) {
    const int offA = kt * 32 + hv * 16;  // A layout: {0-15}/{16-31}
    const int offB = kt * 32 + hv * 16;  // B layout: {0-15}/{16-31}
    FragU fq, fk, fv;
    fq.h[0] = *(const U4*)(wq + offB);
    fq.h[1] = *(const U4*)(wq + offB + 8);
    fk.h[0] = *(const U4*)(wk + offB);
    fk.h[1] = *(const U4*)(wk + offB + 8);
    fv.h[0] = *(const U4*)(wv + offB);
    fv.h[1] = *(const U4*)(wv + offB + 8);
#pragma unroll
    for (int i = 0; i < MB; ++i) {
      FragU a;
      a.h[0] = *(const U4*)(xr[i] + offA);
      a.h[1] = *(const U4*)(xr[i] + offA + 8);
      cq[i] = wmma_bf16(a.v, fq.v, cq[i]);
      ck[i] = wmma_bf16(a.v, fk.v, ck[i]);
      cv[i] = wmma_bf16(a.v, fv.v, cv[i]);
    }
  }

  // Epilogue: bias + RoPE (q,k) + scatter into padded attention layouts.
  const float bqd = bq[d], bkd = bk[d], bvd = bv[d];
  const int hh = d / HDc;
  const int hd = d - hh * HDc;
  const int pr = hd >> 1;
  const bool isEven = (hd & 1) == 0;

#pragma unroll
  for (int i = 0; i < MB; ++i) {
    const int mt = mtb + i;
    float qv[8], kv[8];
#pragma unroll
    for (int r = 0; r < 8; ++r) { qv[r] = cq[i][r] + bqd; kv[r] = ck[i][r] + bkd; }
#pragma unroll
    for (int r = 0; r < 8; ++r) {
      const float qp = __shfl_xor(qv[r], 1, 32);  // RoPE partner (d^1 = lane^1)
      const float kp = __shfl_xor(kv[r], 1, 32);
      if (mt < MT) {
        const int g  = mt * 16 + r + 8 * hv;      // global row = b*S + s
        const int bb = g / Sc;
        const int ss = g - bb * Sc;
        const float c_ = fcos[ss * HD2c + pr];
        const float s_ = fsin[ss * HD2c + pr];
        const float qo = isEven ? (qv[r] * c_ - qp * s_) : (qp * s_ + qv[r] * c_);
        const float ko = isEven ? (kv[r] * c_ - kp * s_) : (kp * s_ + kv[r] * c_);
        const size_t qi = ((size_t)(bb * Hc + hh) * SPc + ss) * HDPc + hd;
        qbuf[qi] = f2bf(qo);
        kbuf[qi] = f2bf(ko);
        vtb[((size_t)(bb * Hc + hh) * HDPc + hd) * SKPc + ss] = f2bf(cv[i][r] + bvd);
      }
    }
  }
}

// ---------------------------------------------------------------------------
// Flash attention: one wave per (b,h, 2x16-query tiles).
// Logits computed transposed (keys = C rows) so exp->A repack is lane-local.
// K A-fragments and V B-fragments are shared across both query tiles.
__global__ __launch_bounds__(32) void attn_kernel(
    const unsigned short* __restrict__ qbuf,
    const unsigned short* __restrict__ kbuf,
    const unsigned short* __restrict__ vtb,
    unsigned short* __restrict__ at)     // bf16 [B,S,D]
{
  const int qt0 = blockIdx.x * QW;       // first query tile
  const int bh  = blockIdx.y;            // 0..255
  const int b   = bh >> 4, h = bh & 15;
  const int lane = threadIdx.x;
  const int hv  = lane >> 4;
  const int lid = lane & 15;
  const float scale = 0.10660035817780521f;   // 1/sqrt(88)

  // Q as B-fragments (K=hd chunks of 32), lane = query index (row-clamped).
  FragU Qf[QW][3];
#pragma unroll
  for (int w = 0; w < QW; ++w) {
    const int qrow = imin((qt0 + w) * 16 + lid, SPc - 1);
    const unsigned short* qr = qbuf + ((size_t)bh * SPc + qrow) * HDPc;
#pragma unroll
    for (int st = 0; st < 3; ++st) {
      const int off = st * 32 + hv * 16;
      Qf[w][st].h[0] = *(const U4*)(qr + off);
      Qf[w][st].h[1] = *(const U4*)(qr + off + 8);
    }
  }

  float mrow[QW], lrow[QW];
  v8f O[QW][6];
#pragma unroll
  for (int w = 0; w < QW; ++w) {
    mrow[w] = -1e30f; lrow[w] = 0.f;
#pragma unroll
    for (int nt = 0; nt < 6; ++nt) O[w][nt] = {};
  }

  for (int kb2 = 0; kb2 < KBLK; ++kb2) {
    v8f stw[QW][2];
#pragma unroll
    for (int sub = 0; sub < 2; ++sub) {
      const int key0 = kb2 * 32 + sub * 16;
      const unsigned short* kr =
          kbuf + ((size_t)bh * SPc + key0 + lid) * HDPc;   // lane = key row
      v8f c[QW];
#pragma unroll
      for (int w = 0; w < QW; ++w) c[w] = {};
#pragma unroll
      for (int st = 0; st < 3; ++st) {
        const int off = st * 32 + hv * 16;
        FragU A;                                   // shared across query tiles
        A.h[0] = *(const U4*)(kr + off);
        A.h[1] = *(const U4*)(kr + off + 8);
#pragma unroll
        for (int w = 0; w < QW; ++w)
          c[w] = wmma_bf16(A.v, Qf[w][st].v, c[w]);        // S^T tiles
      }
#pragma unroll
      for (int w = 0; w < QW; ++w) {
#pragma unroll
        for (int r = 0; r < 8; ++r) {
          const int key = key0 + r + 8 * hv;
          c[w][r] = (key < Sc) ? c[w][r] * scale : -1e30f;
        }
        stw[w][sub] = c[w];
      }
    }

    // Online softmax (per query = lid; lanes L / L+16 hold disjoint keys).
    FragU P[QW];
#pragma unroll
    for (int w = 0; w < QW; ++w) {
      float mtile = -1e30f;
#pragma unroll
      for (int r = 0; r < 8; ++r)
        mtile = fmaxf(mtile, fmaxf(stw[w][0][r], stw[w][1][r]));
      mtile = fmaxf(mtile, __shfl_xor(mtile, 16, 32));
      const float mnew  = fmaxf(mrow[w], mtile);
      const float alpha = __expf(mrow[w] - mnew);

      float psum = 0.f;
#pragma unroll
      for (int j = 0; j < 4; ++j) {
        const float p0 = __expf(stw[w][0][2 * j]     - mnew);
        const float p1 = __expf(stw[w][0][2 * j + 1] - mnew);
        const float p2 = __expf(stw[w][1][2 * j]     - mnew);
        const float p3 = __expf(stw[w][1][2 * j + 1] - mnew);
        P[w].u[j]     = pack2bf(p0, p1);   // A frag K {0-7}/{8-15}
        P[w].u[4 + j] = pack2bf(p2, p3);   // A frag K {16-23}/{24-31}
        psum += (p0 + p1) + (p2 + p3);
      }
      psum += __shfl_xor(psum, 16, 32);
      lrow[w] = lrow[w] * alpha + psum;
      mrow[w] = mnew;

      float arow[8];
#pragma unroll
      for (int r = 0; r < 8; ++r) arow[r] = __shfl(alpha, r + 8 * hv, 32);
#pragma unroll
      for (int nt = 0; nt < 6; ++nt)
#pragma unroll
        for (int r = 0; r < 8; ++r) O[w][nt][r] *= arow[r];
    }

    // O += P @ V   (B = V^T rows: lane = hd, K = keys; shared across tiles)
#pragma unroll
    for (int nt = 0; nt < 6; ++nt) {
      const unsigned short* vr =
          vtb + ((size_t)bh * HDPc + nt * 16 + lid) * SKPc + kb2 * 32 + hv * 8;
      FragU Bv;
      Bv.h[0] = *(const U4*)(vr);
      Bv.h[1] = *(const U4*)(vr + 16);
#pragma unroll
      for (int w = 0; w < QW; ++w)
        O[w][nt] = wmma_bf16(P[w].v, Bv.v, O[w][nt]);
    }
  }

#pragma unroll
  for (int w = 0; w < QW; ++w) {
    const float linv = 1.0f / lrow[w];
    float lr[8];
#pragma unroll
    for (int r = 0; r < 8; ++r) lr[r] = __shfl(linv, r + 8 * hv, 32);
#pragma unroll
    for (int nt = 0; nt < 6; ++nt) {
      const int hd = nt * 16 + lid;
#pragma unroll
      for (int r = 0; r < 8; ++r) {
        const int s = (qt0 + w) * 16 + r + 8 * hv;
        if (s < Sc && hd < HDc)
          at[((size_t)(b * Sc + s)) * Dc + h * HDc + hd] = f2bf(O[w][nt][r] * lr[w == w ? r : r]);
      }
    }
  }
}

// ---------------------------------------------------------------------------
// Output projection: out = attn @ Wo^T + bo (fp32 store), 4x4 tile blocking.
__global__ __launch_bounds__(32) void oproj_kernel(
    const unsigned short* __restrict__ atb,
    const unsigned short* __restrict__ wob,
    const float* __restrict__ bo,
    float* __restrict__ out)
{
  const int ntb = blockIdx.x * ONB;
  const int mtb = blockIdx.y * OMB;
  const int lane = threadIdx.x;
  const int hv  = lane >> 4;
  const int lid = lane & 15;

  const unsigned short* ar[OMB];
#pragma unroll
  for (int i = 0; i < OMB; ++i) {
    const int row = imin((mtb + i) * 16 + lid, Mc - 1);
    ar[i] = atb + (size_t)row * Dc;
  }
  int dcol[ONB];
  const unsigned short* wr[ONB];
#pragma unroll
  for (int j = 0; j < ONB; ++j) {
    dcol[j] = (ntb + j) * 16 + lid;
    wr[j] = wob + (size_t)dcol[j] * Dc;
  }

  v8f c[OMB][ONB];
#pragma unroll
  for (int i = 0; i < OMB; ++i)
#pragma unroll
    for (int j = 0; j < ONB; ++j) c[i][j] = {};

  for (int kt = 0; kt < KT; ++kt) {
    const int offA = kt * 32 + hv * 16;
    const int offB = kt * 32 + hv * 16;
    FragU bm[ONB];
#pragma unroll
    for (int j = 0; j < ONB; ++j) {
      bm[j].h[0] = *(const U4*)(wr[j] + offB);
      bm[j].h[1] = *(const U4*)(wr[j] + offB + 8);
    }
#pragma unroll
    for (int i = 0; i < OMB; ++i) {
      FragU a;
      a.h[0] = *(const U4*)(ar[i] + offA);
      a.h[1] = *(const U4*)(ar[i] + offA + 8);
#pragma unroll
      for (int j = 0; j < ONB; ++j)
        c[i][j] = wmma_bf16(a.v, bm[j].v, c[i][j]);
    }
  }

#pragma unroll
  for (int j = 0; j < ONB; ++j) {
    const float bod = bo[dcol[j]];
#pragma unroll
    for (int i = 0; i < OMB; ++i) {
      const int mt = mtb + i;
      if (mt < MT) {
#pragma unroll
        for (int r = 0; r < 8; ++r) {
          const int g = mt * 16 + r + 8 * hv;   // row over B*S
          out[(size_t)g * Dc + dcol[j]] = c[i][j][r] + bod;
        }
      }
    }
  }
}

// ---------------------------------------------------------------------------
extern "C" void kernel_launch(void* const* d_in, const int* in_sizes, int n_in,
                              void* d_out, int out_size, void* d_ws,
                              size_t ws_size, hipStream_t stream) {
  (void)in_sizes; (void)n_in; (void)out_size; (void)ws_size;

  const float* x    = (const float*)d_in[0];
  const float* fcos = (const float*)d_in[1];
  const float* fsin = (const float*)d_in[2];
  const float* Wq   = (const float*)d_in[3];
  const float* bq   = (const float*)d_in[4];
  const float* Wk   = (const float*)d_in[5];
  const float* bk   = (const float*)d_in[6];
  const float* Wv   = (const float*)d_in[7];
  const float* bv   = (const float*)d_in[8];
  const float* Wo   = (const float*)d_in[9];
  const float* bo   = (const float*)d_in[10];

  char* ws = (char*)d_ws;
  size_t off = 0;
  unsigned short* xb  = (unsigned short*)(ws + off); off += XB_BYTES;
  unsigned short* wqb = (unsigned short*)(ws + off); off += W_BYTES;
  unsigned short* wkb = (unsigned short*)(ws + off); off += W_BYTES;
  unsigned short* wvb = (unsigned short*)(ws + off); off += W_BYTES;
  unsigned short* wob = (unsigned short*)(ws + off); off += W_BYTES;
  unsigned short* qbf = (unsigned short*)(ws + off); off += QKB_BYTES;
  unsigned short* kbf = (unsigned short*)(ws + off); off += QKB_BYTES;
  unsigned short* vtb = (unsigned short*)(ws + off); off += VT_BYTES;
  unsigned short* atb = (unsigned short*)(ws + off); off += AT_BYTES;

  const int nX4 = Mc * Dc / 4, nW4 = Dc * Dc / 4;
  cvt_f32_bf16_x4<<<(nX4 + 255) / 256, 256, 0, stream>>>(
      (const float4*)x, (uint2*)xb, nX4);
  cvt_f32_bf16_x4<<<(nW4 + 255) / 256, 256, 0, stream>>>(
      (const float4*)Wq, (uint2*)wqb, nW4);
  cvt_f32_bf16_x4<<<(nW4 + 255) / 256, 256, 0, stream>>>(
      (const float4*)Wk, (uint2*)wkb, nW4);
  cvt_f32_bf16_x4<<<(nW4 + 255) / 256, 256, 0, stream>>>(
      (const float4*)Wv, (uint2*)wvb, nW4);
  cvt_f32_bf16_x4<<<(nW4 + 255) / 256, 256, 0, stream>>>(
      (const float4*)Wo, (uint2*)wob, nW4);

  // Zero padded regions (graph-capture-safe async memsets).
  hipMemsetAsync(qbf, 0, QKB_BYTES, stream);
  hipMemsetAsync(kbf, 0, QKB_BYTES, stream);
  hipMemsetAsync(vtb, 0, VT_BYTES, stream);

  qkv_rope_kernel<<<dim3(NT, MBLK), 32, 0, stream>>>(
      xb, wqb, wkb, wvb, bq, bk, bv, fcos, fsin, qbf, kbf, vtb);

  attn_kernel<<<dim3(QBLK, Bc * Hc), 32, 0, stream>>>(qbf, kbf, vtb, atb);

  oproj_kernel<<<dim3(ONBLK, OMBLK), 32, 0, stream>>>(atb, wob, bo,
                                                      (float*)d_out);
}
